// TriangleMultiplicativeUpdate_4629974745259
// MI455X (gfx1250) — matrix operand order repaired and checked by reading, not compiled, not run on hardware
//
#include <hip/hip_runtime.h>
#include <hip/hip_bf16.h>

// ---------------------------------------------------------------------------
// Triangle multiplicative update (outgoing), MI455X / gfx1250.
// All GEMM math through v_wmma_f32_16x16x32_bf16 (bf16 in, fp32 accumulate).
// Intermediates kept bf16 + h-major so the O(N^3 H) einsum is pure contiguous
// WMMA-fragment traffic that lives in the 192MB L2.
// ---------------------------------------------------------------------------

#define NDIM   384
#define DDIM   128
#define HDIM   128
#define M_ROWS (NDIM * NDIM)          // 147456 flattened (i,j)/(i,k) rows
#define EPSV   1e-5f
#define FRAG_MAT (4 * 8 * 32 * 16)    // per-matrix swizzled weight elems (16384)

typedef __attribute__((ext_vector_type(16))) __bf16 bf16x16;
typedef __attribute__((ext_vector_type(8)))  float  floatx8;

#define WMMA_BF16(a, b, c) \
  __builtin_amdgcn_wmma_f32_16x16x32_bf16(false, (a), false, (b), (short)0, (c), false, false)

__device__ __forceinline__ float sigmoidf(float v) { return 1.0f / (1.0f + __expf(-v)); }

__device__ __forceinline__ unsigned short bfbits(float f) {
  union { __bf16 h; unsigned short u; } c; c.h = (__bf16)f; return c.u;
}

// A-fragment (16x32 bf16) gather from an LDS tile laid out [16 rows][128 k].
// Layout: lanes 0-15 row M=lane, K={koff..koff+15} (contiguous, matches B frags),
// koff = (lane>>4)*16.  Two ds_load_b128 per fragment.
__device__ __forceinline__ bf16x16 afrag_lds(const __bf16* base, int kt) {
  const int lane = threadIdx.x & 31;
  const __bf16* row = base + (lane & 15) * 128 + kt * 32;
  const int koff = (lane >> 4) * 16;
  union { bf16x16 v; uint4 q[2]; } r;
  r.q[0] = *(const uint4*)(row + koff);
  r.q[1] = *(const uint4*)(row + 8 + koff);
  return r.v;
}

// B-fragment from pre-swizzled weights: one contiguous 32B load per lane.
__device__ __forceinline__ bf16x16 bfrag_w(const __bf16* frag, int kt, int nt) {
  const int lane = threadIdx.x & 31;
  return *(const bf16x16*)(frag + (size_t)(((kt * 8 + nt) * 32) + lane) * 16);
}

// ---------------------------------------------------------------------------
// Kernel 0: swizzle the six 128x128 fp32 weight matrices into contiguous-k
// B-fragment order (bf16): frag[(m,kt,nt,lane,e)] = W[kt*32+(lane>>4)*16+e][nt*16+(lane&15)]
// ---------------------------------------------------------------------------
__global__ __launch_bounds__(256) void prep_weights(
    const float* __restrict__ Wl, const float* __restrict__ Wlg,
    const float* __restrict__ Wr, const float* __restrict__ Wrg,
    const float* __restrict__ Wout, const float* __restrict__ Wg,
    __bf16* __restrict__ frags) {
  const int t = blockIdx.x * 256 + threadIdx.x;
  if (t >= 6 * FRAG_MAT) return;
  const int e    = t & 15;
  const int lane = (t >> 4) & 31;
  const int nt   = (t >> 9) & 7;
  const int kt   = (t >> 12) & 3;
  const int m    = t >> 14;
  const float* W = (m == 0) ? Wl : (m == 1) ? Wlg : (m == 2) ? Wr
                 : (m == 3) ? Wrg : (m == 4) ? Wout : Wg;
  const int n = nt * 16 + (lane & 15);
  const int k = kt * 32 + (lane >> 4) * 16 + e;
  frags[t] = (__bf16)W[k * 128 + n];
}

// ---------------------------------------------------------------------------
// Kernel 1: fused LayerNorm(x) + 4 projections + sigmoid gating.
// One block = 16 rows of the flattened [147456 x 128] problem, 4 waves.
// Writes left/right h-major: buf[h][i][k] (bf16) for the einsum.
// ---------------------------------------------------------------------------
__global__ __launch_bounds__(128) void proj_kernel(
    const float* __restrict__ x, const float* __restrict__ ng, const float* __restrict__ nb,
    const __bf16* __restrict__ frags,
    const float* __restrict__ bl, const float* __restrict__ blg,
    const float* __restrict__ br, const float* __restrict__ brg,
    __bf16* __restrict__ left_t, __bf16* __restrict__ right_t) {
  __shared__ __bf16 zsm[16 * 128];
  const int tid  = threadIdx.x;
  const int row0 = blockIdx.x * 16;
  const int irow = row0 / NDIM;   // 16 | 384, so one i per tile
  const int k0   = row0 % NDIM;

  { // ---- LayerNorm over D=128: 8 lanes per row, 16 elems per lane ----
    const int m   = tid >> 3;
    const int seg = (tid & 7) * 16;
    const float* xr = x + (size_t)(row0 + m) * DDIM + seg;
    float v[16], s = 0.f, s2 = 0.f;
#pragma unroll
    for (int j = 0; j < 4; ++j) {
      float4 f = ((const float4*)xr)[j];
      v[4 * j + 0] = f.x; v[4 * j + 1] = f.y; v[4 * j + 2] = f.z; v[4 * j + 3] = f.w;
      s  += f.x + f.y + f.z + f.w;
      s2 += f.x * f.x + f.y * f.y + f.z * f.z + f.w * f.w;
    }
#pragma unroll
    for (int o = 1; o < 8; o <<= 1) { s += __shfl_xor(s, o, 32); s2 += __shfl_xor(s2, o, 32); }
    const float mu   = s * (1.0f / DDIM);
    const float rstd = rsqrtf(s2 * (1.0f / DDIM) - mu * mu + EPSV);
#pragma unroll
    for (int j = 0; j < 16; ++j)
      zsm[m * 128 + seg + j] = (__bf16)((v[j] - mu) * rstd * ng[seg + j] + nb[seg + j]);
  }
  __syncthreads();

  const int w = tid >> 5, lane = tid & 31;
  bf16x16 A[4];
#pragma unroll
  for (int kt = 0; kt < 4; ++kt) A[kt] = afrag_lds(zsm, kt);

#pragma unroll
  for (int side = 0; side < 2; ++side) {           // 0: left (Wl,Wlg)  1: right (Wr,Wrg)
    const __bf16* fv = frags + (size_t)(side ? 2 : 0) * FRAG_MAT;
    const __bf16* fg = fv + FRAG_MAT;
    const float*  bv = side ? br : bl;
    const float*  bg = side ? brg : blg;
    __bf16* dst = side ? right_t : left_t;
#pragma unroll
    for (int tt = 0; tt < 2; ++tt) {               // each wave owns n-tiles {w, w+4}
      const int nt = w + 4 * tt;
      floatx8 av = {0, 0, 0, 0, 0, 0, 0, 0};
      floatx8 ag = {0, 0, 0, 0, 0, 0, 0, 0};
#pragma unroll
      for (int kt = 0; kt < 4; ++kt) {
        av = WMMA_BF16(A[kt], bfrag_w(fv, kt, nt), av);
        ag = WMMA_BF16(A[kt], bfrag_w(fg, kt, nt), ag);
      }
      const int h = nt * 16 + (lane & 15);
      const float bvv = bv[h], bgv = bg[h];
      unsigned p[4];
#pragma unroll
      for (int r = 0; r < 4; ++r) {                // pack 8 k-consecutive rows -> b128 store
        const float lo = (av[2 * r]     + bvv) * sigmoidf(ag[2 * r]     + bgv);
        const float hi = (av[2 * r + 1] + bvv) * sigmoidf(ag[2 * r + 1] + bgv);
        p[r] = (unsigned)bfbits(lo) | ((unsigned)bfbits(hi) << 16);
      }
      uint4 st; st.x = p[0]; st.y = p[1]; st.z = p[2]; st.w = p[3];
      const size_t doff = ((size_t)h * NDIM + irow) * NDIM + k0 + (lane >> 4) * 8;
      *(uint4*)(dst + doff) = st;
    }
  }
}

// ---------------------------------------------------------------------------
// Kernel 2: core[i,j,h] = sum_k left[i,k,h]*right[j,k,h].
// Block = 16x16 (i,j) tile x all 128 h; 8 waves, 16 h-slices each.
// Per h: 12 x v_wmma_f32_16x16x32_bf16 over K=384, all operand loads
// contiguous b128/b256 thanks to the h-major bf16 layout (L2-resident).
// ---------------------------------------------------------------------------
__global__ __launch_bounds__(256) void einsum_kernel(
    const __bf16* __restrict__ left_t, const __bf16* __restrict__ right_t,
    __bf16* __restrict__ core) {
  const int i0 = blockIdx.x * 16, j0 = blockIdx.y * 16;
  const int tid = threadIdx.x, w = tid >> 5, lane = tid & 31;
  const int sel  = lane & 15;       // A: row M, B: col N
  const int hb   = lane >> 4;
  const int koffA = hb * 16, koffB = hb * 16;

  for (int hh = 0; hh < 16; ++hh) {
    const int h = w * 16 + hh;
    const __bf16* Lh = left_t  + (size_t)h * M_ROWS + (size_t)(i0 + sel) * NDIM;
    const __bf16* Rh = right_t + (size_t)h * M_ROWS + (size_t)(j0 + sel) * NDIM;
    floatx8 acc = {0, 0, 0, 0, 0, 0, 0, 0};
#pragma unroll 4
    for (int kt = 0; kt < 12; ++kt) {
      union { bf16x16 v; uint4 q[2]; } a;
      a.q[0] = *(const uint4*)(Lh + kt * 32 + koffA);
      a.q[1] = *(const uint4*)(Lh + kt * 32 + 8 + koffA);
      const bf16x16 b = *(const bf16x16*)(Rh + kt * 32 + koffB);
      acc = WMMA_BF16(a.v, b, acc);
    }
    const int j = j0 + sel;
#pragma unroll
    for (int r = 0; r < 8; ++r) {
      const int ii = i0 + r + 8 * hb;
      core[((size_t)ii * NDIM + j) * HDIM + h] = (__bf16)acc[r];
    }
  }
}

// ---------------------------------------------------------------------------
// Kernel 3: fused LayerNorm(core) @ Wout + bout, gate sigmoid(x@Wg+bg),
// residual: out = x + g * zo.  Same 16-row / 4-wave structure as kernel 1.
// ---------------------------------------------------------------------------
__global__ __launch_bounds__(128) void out_kernel(
    const __bf16* __restrict__ core, const float* __restrict__ x,
    const float* __restrict__ og, const float* __restrict__ ob,
    const __bf16* __restrict__ frags,
    const float* __restrict__ bout, const float* __restrict__ bgv_,
    float* __restrict__ out) {
  __shared__ __bf16 zo[16 * 128];
  __shared__ __bf16 xb[16 * 128];
  __shared__ float  xs[16 * 128];
  const int tid = threadIdx.x;
  const int row0 = blockIdx.x * 16;

  {
    const int m = tid >> 3, seg = (tid & 7) * 16;
    // LayerNorm of core over H
    const __bf16* cr = core + (size_t)(row0 + m) * HDIM + seg;
    union { bf16x16 v; uint4 q[2]; } cl;
    cl.q[0] = *(const uint4*)(cr);
    cl.q[1] = *(const uint4*)(cr + 8);
    float v[16], s = 0.f, s2 = 0.f;
#pragma unroll
    for (int j = 0; j < 16; ++j) { v[j] = (float)cl.v[j]; s += v[j]; s2 += v[j] * v[j]; }
#pragma unroll
    for (int o = 1; o < 8; o <<= 1) { s += __shfl_xor(s, o, 32); s2 += __shfl_xor(s2, o, 32); }
    const float mu = s * (1.0f / HDIM);
    const float rstd = rsqrtf(s2 * (1.0f / HDIM) - mu * mu + EPSV);
#pragma unroll
    for (int j = 0; j < 16; ++j)
      zo[m * 128 + seg + j] = (__bf16)((v[j] - mu) * rstd * og[seg + j] + ob[seg + j]);
    // stage x tile (fp32 for residual, bf16 for the gate GEMM)
    const float* xr = x + (size_t)(row0 + m) * DDIM + seg;
#pragma unroll
    for (int j = 0; j < 4; ++j) {
      float4 f = ((const float4*)xr)[j];
      xs[m * 128 + seg + 4 * j + 0] = f.x; xb[m * 128 + seg + 4 * j + 0] = (__bf16)f.x;
      xs[m * 128 + seg + 4 * j + 1] = f.y; xb[m * 128 + seg + 4 * j + 1] = (__bf16)f.y;
      xs[m * 128 + seg + 4 * j + 2] = f.z; xb[m * 128 + seg + 4 * j + 2] = (__bf16)f.z;
      xs[m * 128 + seg + 4 * j + 3] = f.w; xb[m * 128 + seg + 4 * j + 3] = (__bf16)f.w;
    }
  }
  __syncthreads();

  const int w = tid >> 5, lane = tid & 31;
  bf16x16 Ao[4], Ax[4];
#pragma unroll
  for (int kt = 0; kt < 4; ++kt) { Ao[kt] = afrag_lds(zo, kt); Ax[kt] = afrag_lds(xb, kt); }
  const __bf16* fo = frags + (size_t)4 * FRAG_MAT;   // Wout
  const __bf16* fg = frags + (size_t)5 * FRAG_MAT;   // Wg

#pragma unroll
  for (int tt = 0; tt < 2; ++tt) {
    const int nt = w + 4 * tt;
    floatx8 ao = {0, 0, 0, 0, 0, 0, 0, 0};
    floatx8 agt = {0, 0, 0, 0, 0, 0, 0, 0};
#pragma unroll
    for (int kt = 0; kt < 4; ++kt) {
      ao  = WMMA_BF16(Ao[kt], bfrag_w(fo, kt, nt), ao);
      agt = WMMA_BF16(Ax[kt], bfrag_w(fg, kt, nt), agt);
    }
    const int d = nt * 16 + (lane & 15);
    const float bo = bout[d], bgg = bgv_[d];
    const int hb = lane >> 4;
#pragma unroll
    for (int r = 0; r < 8; ++r) {
      const int m = r + 8 * hb;
      out[(size_t)(row0 + m) * DDIM + d] =
          xs[m * 128 + d] + sigmoidf(agt[r] + bgg) * (ao[r] + bo);
    }
  }
}

// ---------------------------------------------------------------------------
extern "C" void kernel_launch(void* const* d_in, const int* in_sizes, int n_in,
                              void* d_out, int out_size, void* d_ws, size_t ws_size,
                              hipStream_t stream) {
  (void)in_sizes; (void)n_in; (void)out_size; (void)ws_size;
  const float* x    = (const float*)d_in[0];
  const float* ng   = (const float*)d_in[1];
  const float* nb   = (const float*)d_in[2];
  const float* Wl   = (const float*)d_in[3];
  const float* bl   = (const float*)d_in[4];
  const float* Wlg  = (const float*)d_in[5];
  const float* blg  = (const float*)d_in[6];
  const float* Wr   = (const float*)d_in[7];
  const float* br   = (const float*)d_in[8];
  const float* Wrg  = (const float*)d_in[9];
  const float* brg  = (const float*)d_in[10];
  const float* og   = (const float*)d_in[11];
  const float* ob   = (const float*)d_in[12];
  const float* Wout = (const float*)d_in[13];
  const float* bout = (const float*)d_in[14];
  const float* Wg   = (const float*)d_in[15];
  const float* bg   = (const float*)d_in[16];
  float* out = (float*)d_out;

  char* ws = (char*)d_ws;
  const size_t HALF_BUF = (size_t)HDIM * M_ROWS * sizeof(__bf16);   // 37,748,736 B
  __bf16* frags   = (__bf16*)ws;                                    // 196,608 B
  __bf16* left_t  = (__bf16*)(ws + 196608);
  __bf16* right_t = (__bf16*)(ws + 196608 + HALF_BUF);
  __bf16* core    = (__bf16*)(ws + 196608 + 2 * HALF_BUF);

  prep_weights<<<(6 * FRAG_MAT + 255) / 256, 256, 0, stream>>>(Wl, Wlg, Wr, Wrg, Wout, Wg, frags);
  proj_kernel<<<M_ROWS / 16, 128, 0, stream>>>(x, ng, nb, frags, bl, blg, br, brg, left_t, right_t);
  einsum_kernel<<<dim3(NDIM / 16, NDIM / 16), 256, 0, stream>>>(left_t, right_t, core);
  out_kernel<<<M_ROWS / 16, 128, 0, stream>>>(core, x, og, ob, frags, bout, bg, out);
}
